// NLgate_1511828489046
// MI455X (gfx1250) — matrix operand order repaired and checked by reading, not compiled, not run on hardware
//
#include <hip/hip_runtime.h>
#include <hip/hip_bf16.h>

// ---------------------------------------------------------------------------
// Problem constants (from reference): B=16, THW=TF=2048, C=DQ=DK=512
// ---------------------------------------------------------------------------
#define BATCH 16
#define TSEQ  2048
#define CDIM  512
#define MROWS (BATCH * TSEQ)   // 32768

typedef __attribute__((ext_vector_type(16))) __bf16 v16bf;
typedef __attribute__((ext_vector_type(8)))  float  v8f;
typedef __attribute__((ext_vector_type(4)))  unsigned int v4u;
typedef __attribute__((ext_vector_type(4)))  int    v4i;
typedef __attribute__((ext_vector_type(8)))  int    v8i;

// ---------------------------------------------------------------------------
// Tensor Data Mover: 2D tile load Global->LDS (CDNA5 ISA)
// D# group0: [1:0]=count=1, [63:32]=lds byte addr, [120:64]=global addr,
//            [127:126]=type=2.
// D# group1: [17:16]=data_size (1 => 2 bytes), [79:48]=tensor_dim0,
//            [111:80]=tensor_dim1, [127:112]=tile_dim0, [143:128]=tile_dim1,
//            [207:160]=tensor_dim0_stride. groups 2/3 zero (2D tile).
// Completion tracked with TENSORcnt; per-wave completions are IN ORDER, so a
// wave that keeps <=1 outstanding (s_wait_tensorcnt 1) knows all older loads
// have landed -> enables double-buffered pipelining.
// ---------------------------------------------------------------------------
__device__ __forceinline__ void tdm_load_2d(unsigned lds_addr, const void* gaddr,
                                            unsigned tensor_d0, unsigned tensor_d1,
                                            unsigned tile_d0, unsigned tile_d1,
                                            unsigned stride_d0_elems) {
  unsigned long long ga = (unsigned long long)(size_t)gaddr;
  v4u g0;
  g0[0] = 1u;                                         // count=1 (user mode)
  g0[1] = lds_addr;                                   // LDS byte address
  g0[2] = (unsigned)(ga & 0xffffffffu);               // global_addr[31:0]
  g0[3] = (unsigned)((ga >> 32) & 0x01ffffffu) | (2u << 30);  // addr[56:32]|type=2
  v8i g1;
  g1[0] = (int)(1u << 16);                            // data_size = 2 bytes
  g1[1] = (int)((tensor_d0 & 0xffffu) << 16);         // tensor_dim0[15:0]
  g1[2] = (int)(((tensor_d0 >> 16) & 0xffffu) | ((tensor_d1 & 0xffffu) << 16));
  g1[3] = (int)(((tensor_d1 >> 16) & 0xffffu) | ((tile_d0 & 0xffffu) << 16));
  g1[4] = (int)(tile_d1 & 0xffffu);                   // tile_dim1 (tile_dim2=0)
  g1[5] = (int)stride_d0_elems;                       // tensor_dim0_stride[31:0]
  g1[6] = 0;
  g1[7] = 0;
  v4i z4 = {};
#if defined(__clang_major__) && (__clang_major__ >= 23)
  v8i z8 = {};
  __builtin_amdgcn_tensor_load_to_lds(g0, g1, z4, z4, z8, 0);
#else
  __builtin_amdgcn_tensor_load_to_lds(g0, g1, z4, z4, 0);
#endif
}

__device__ __forceinline__ unsigned lds_off(const void* p) {
  return (unsigned)(size_t)p;   // low 32 bits of generic ptr = LDS byte offset
}

// ---------------------------------------------------------------------------
// WMMA fragment helpers (wave32 layouts, CDNA5 ISA §7.12.2)
// ---------------------------------------------------------------------------
__device__ __forceinline__ v16bf frag_a(const __bf16* lds, int stride,
                                        int m0, int k0, int lane) {
  int row = m0 + (lane & 15);
  int off = (lane & 16) ? 8 : 0;
  union { v16bf v; uint4 q[2]; } u;
  const __bf16* p = lds + (size_t)row * stride + k0 + off;
  u.q[0] = *(const uint4*)(p);
  u.q[1] = *(const uint4*)(p + 16);
  return u.v;
}

__device__ __forceinline__ v16bf frag_b(const __bf16* ldsT, int stride,
                                        int n0, int k0, int lane) {
  int col = n0 + (lane & 15);
  int off = (lane & 16) ? 8 : 0;
  union { v16bf v; uint4 q[2]; } u;
  const __bf16* p = ldsT + (size_t)col * stride + k0 + off;
  u.q[0] = *(const uint4*)(p);
  u.q[1] = *(const uint4*)(p + 16);
  return u.v;
}

__device__ __forceinline__ v8f wmma_bf16(v16bf a, v16bf b, v8f c) {
  return __builtin_amdgcn_wmma_f32_16x16x32_bf16(false, a, false, b,
                                                 (short)0, c, false, false);
}

// ---------------------------------------------------------------------------
// Kernel 0a: f32 -> bf16 bulk convert (activations), 8 elems / thread.
// ---------------------------------------------------------------------------
__global__ __launch_bounds__(256)
void cvt_kernel(const float* __restrict__ src, __bf16* __restrict__ dst) {
  size_t i = ((size_t)blockIdx.x * 256 + threadIdx.x) * 8;
  float4 a = *(const float4*)(src + i);
  float4 b = *(const float4*)(src + i + 4);
  __align__(16) __bf16 t[8] = {(__bf16)a.x, (__bf16)a.y, (__bf16)a.z, (__bf16)a.w,
                               (__bf16)b.x, (__bf16)b.y, (__bf16)b.z, (__bf16)b.w};
  *(uint4*)(dst + i) = *(const uint4*)t;
}

// ---------------------------------------------------------------------------
// Kernel 0b: weight transpose + convert: wT[n][k] = (bf16) W[k][n]
// ---------------------------------------------------------------------------
__global__ __launch_bounds__(256)
void wt_kernel(const float* __restrict__ W, __bf16* __restrict__ wT) {
  int idx = blockIdx.x * 256 + threadIdx.x;
  int kk = idx >> 9;
  int n  = idx & 511;
  wT[(size_t)n * CDIM + kk] = (__bf16)W[(size_t)kk * CDIM + n];
}

// ---------------------------------------------------------------------------
// Kernel 1: projection GEMM  Y[M,512] = Xb[M,512] @ wT^T + bias
//   mode 0 (q): natural bf16 out + f32 out (qin -> d_out)
//   mode 1 (k): scatter to kinT[b][t][c]  where kin[c][t]=kflat[c*TSEQ+t]
//   mode 2 (v): scatter to vT[b][c][t]
// Double-buffered TDM staging: load(k+1) overlaps WMMA(k).
// ---------------------------------------------------------------------------
__global__ __launch_bounds__(128)
void proj_kernel(const __bf16* __restrict__ Xb, const __bf16* __restrict__ WT,
                 const float* __restrict__ bias,
                 __bf16* __restrict__ outb, float* __restrict__ outf, int mode) {
  __shared__ __align__(16) __bf16 sA[2][64][32];
  __shared__ __align__(16) __bf16 sBT[2][64][32];

  const int m0   = blockIdx.x * 64;
  const int n0   = blockIdx.y * 64;
  const int tid  = threadIdx.x;
  const int lane = tid & 31;
  const int wave = tid >> 5;
  const int wm   = (wave >> 1) * 32;
  const int wn   = (wave & 1) * 32;

  v8f acc[2][2] = {};

  // Prologue: stage k-step 0 into buffer 0.
  if (wave == 0)
    tdm_load_2d(lds_off(&sA[0][0][0]),  Xb + (size_t)m0 * CDIM,
                CDIM, MROWS, 32, 64, CDIM);
  if (wave == 1)
    tdm_load_2d(lds_off(&sBT[0][0][0]), WT + (size_t)n0 * CDIM,
                CDIM, CDIM, 32, 64, CDIM);

  const int NK = CDIM / 32;   // 16 k-steps
  for (int ks = 0; ks < NK; ++ks) {
    const int cur = ks & 1;
    if (ks + 1 < NK) {
      const int nxt = cur ^ 1;
      if (wave == 0)
        tdm_load_2d(lds_off(&sA[nxt][0][0]),  Xb + (size_t)m0 * CDIM + (ks + 1) * 32,
                    CDIM, MROWS, 32, 64, CDIM);
      if (wave == 1)
        tdm_load_2d(lds_off(&sBT[nxt][0][0]), WT + (size_t)n0 * CDIM + (ks + 1) * 32,
                    CDIM, CDIM, 32, 64, CDIM);
      __builtin_amdgcn_s_wait_tensorcnt((short)1);  // tile ks landed (in-order)
    } else {
      __builtin_amdgcn_s_wait_tensorcnt((short)0);
    }
    __syncthreads();   // tile ks visible to all waves
#pragma unroll
    for (int mi = 0; mi < 2; ++mi) {
      v16bf a = frag_a(&sA[cur][0][0], 32, wm + mi * 16, 0, lane);
#pragma unroll
      for (int ni = 0; ni < 2; ++ni) {
        v16bf b = frag_b(&sBT[cur][0][0], 32, wn + ni * 16, 0, lane);
        acc[mi][ni] = wmma_bf16(a, b, acc[mi][ni]);
      }
    }
    __syncthreads();   // consumers done; next iter may overwrite other buffer
  }

  const int n  = lane & 15;
  const int mb = (lane & 16) ? 8 : 0;
#pragma unroll
  for (int mi = 0; mi < 2; ++mi) {
#pragma unroll
    for (int ni = 0; ni < 2; ++ni) {
      int col = n0 + wn + ni * 16 + n;
      float bv = bias[col];
#pragma unroll
      for (int r = 0; r < 8; ++r) {
        int row = m0 + wm + mi * 16 + mb + r;
        float val = acc[mi][ni][r] + bv;
        if (mode == 0) {
          outb[(size_t)row * CDIM + col] = (__bf16)val;
          outf[(size_t)row * CDIM + col] = val;
        } else if (mode == 1) {
          // raw-reshape transpose: f = lr*512+col within batch; kinT[t][c]
          int bb = row >> 11, lr = row & (TSEQ - 1);
          unsigned f = (unsigned)lr * CDIM + (unsigned)col;
          unsigned c = f >> 11, t = f & (TSEQ - 1);
          outb[(size_t)bb * TSEQ * CDIM + (size_t)t * CDIM + c] = (__bf16)val;
        } else {
          // vT[c][t] = vproj[t][c]
          int bb = row >> 11, lr = row & (TSEQ - 1);
          outb[(size_t)bb * TSEQ * CDIM + (size_t)col * TSEQ + lr] = (__bf16)val;
        }
      }
    }
  }
}

// ---------------------------------------------------------------------------
// Kernel 2: fused flash attention, online softmax.
// Block = (batch, 16 query rows), 128 threads / 4 waves, key tile TK=64.
// Double-buffered TDM staging (wave0: Q then K tiles, wave1: V tiles):
//   sQ     <- qbf  rows q0..q0+15             (tile 512 x 16,  stride 512)
//   sK[i]  <- kinT rows t0..t0+63             (tile 512 x 64,  stride 512)  [t][c]
//   sVT[i] <- vT   cols t0..t0+63, 512 rows   (tile  64 x 512, stride 2048) [c][t]
// Load of tile i+1 overlaps compute on tile i; per-wave in-order TDM
// completion + s_wait_tensorcnt(1) guarantees tile i (and Q) is resident.
// Each wave owns a 16x128 f32 O chunk (8 WMMA C-tiles = 64 VGPRs).
// LDS ~278 KB (within CDNA5's 320 KB per work-group).
// ---------------------------------------------------------------------------
#define TK 64

__global__ __launch_bounds__(128)
void attn_kernel(const __bf16* __restrict__ qbf, const __bf16* __restrict__ kinT,
                 const __bf16* __restrict__ vT, float* __restrict__ io) {
  __shared__ __align__(16) __bf16 sQ[16][CDIM];      // 16 KB
  __shared__ __align__(16) __bf16 sK[2][TK][CDIM];   // 128 KB
  __shared__ __align__(16) __bf16 sVT[2][CDIM][TK];  // 128 KB
  __shared__ __align__(16) float  sS[16][TK];        // 4 KB
  __shared__ __align__(16) __bf16 sP[16][TK];        // 2 KB
  __shared__ float sRed[16][8];
  __shared__ float sM[16], sL[16], sScale[16];

  const int b    = blockIdx.y;
  const int q0   = blockIdx.x * 16;
  const int tid  = threadIdx.x;
  const int lane = tid & 31;
  const int wave = tid >> 5;
  const int cw   = wave * 128;

  const __bf16* qb    = qbf  + ((size_t)b * TSEQ + q0) * CDIM;
  const __bf16* kbase = kinT + (size_t)b * TSEQ * CDIM;
  const __bf16* vbase = vT   + (size_t)b * TSEQ * CDIM;

  // Prologue: wave0 stages Q then K tile 0 (in-order => covered by wait(1));
  // wave1 stages V tile 0.
  if (wave == 0) {
    tdm_load_2d(lds_off(&sQ[0][0]), qb, CDIM, TSEQ, CDIM, 16, CDIM);
    tdm_load_2d(lds_off(&sK[0][0][0]), kbase, CDIM, TSEQ, CDIM, TK, CDIM);
  }
  if (wave == 1)
    tdm_load_2d(lds_off(&sVT[0][0][0]), vbase, TSEQ, CDIM, TK, CDIM, TSEQ);
  if (tid < 16) { sM[tid] = -3.0e38f; sL[tid] = 0.0f; }

  v8f accO[8] = {};

  const int NIT = TSEQ / TK;   // 32 key tiles
  for (int it = 0; it < NIT; ++it) {
    const int cur = it & 1;
    if (it + 1 < NIT) {
      const int nxt = cur ^ 1;
      const int t1 = (it + 1) * TK;
      if (wave == 0)
        tdm_load_2d(lds_off(&sK[nxt][0][0]),  kbase + (size_t)t1 * CDIM,
                    CDIM, TSEQ, CDIM, TK, CDIM);
      if (wave == 1)
        tdm_load_2d(lds_off(&sVT[nxt][0][0]), vbase + t1,
                    TSEQ, CDIM, TK, CDIM, TSEQ);
      __builtin_amdgcn_s_wait_tensorcnt((short)1);  // tile `it` (and Q) landed
    } else {
      __builtin_amdgcn_s_wait_tensorcnt((short)0);
    }
    __syncthreads();

    // Scores: wave w -> S[:, w*16 .. w*16+16) over K = 512 (16 WMMAs).
    {
      v8f s = {};
      const int nb = wave * 16;
      for (int k0 = 0; k0 < CDIM; k0 += 32) {
        v16bf a  = frag_a(&sQ[0][0], CDIM, 0, k0, lane);
        v16bf bf = frag_b(&sK[cur][0][0], CDIM, nb, k0, lane);   // [N=t][K=c]
        s = wmma_bf16(a, bf, s);
      }
      const int n = lane & 15, mb = (lane & 16) ? 8 : 0;
#pragma unroll
      for (int r = 0; r < 8; ++r) sS[mb + r][nb + n] = s[r];
    }
    __syncthreads();

    // Parallel online softmax: 128 threads = 16 rows x 8 segments of 8.
    const int row = tid >> 3, seg = tid & 7;
    {
      float mt = -3.0e38f;
#pragma unroll
      for (int j = 0; j < 8; ++j) mt = fmaxf(mt, sS[row][seg * 8 + j]);
      sRed[row][seg] = mt;
    }
    __syncthreads();
    if (tid < 16) {
      float m = sM[tid];
#pragma unroll
      for (int s2 = 0; s2 < 8; ++s2) m = fmaxf(m, sRed[tid][s2]);
      sScale[tid] = __expf(sM[tid] - m);
      sM[tid] = m;
    }
    __syncthreads();
    {
      float m = sM[row];
      float part = 0.0f;
#pragma unroll
      for (int j = 0; j < 8; ++j) {
        float e = __expf(sS[row][seg * 8 + j] - m);
        sP[row][seg * 8 + j] = (__bf16)e;
        part += e;
      }
      sRed[row][seg] = part;
    }
    __syncthreads();
    if (tid < 16) {
      float l = sL[tid] * sScale[tid];
#pragma unroll
      for (int s2 = 0; s2 < 8; ++s2) l += sRed[tid][s2];
      sL[tid] = l;
    }
    __syncthreads();

    // Rescale O, then O += P @ V  (16 WMMAs per wave).
    {
      const int mb = (lane & 16) ? 8 : 0;
#pragma unroll
      for (int ti = 0; ti < 8; ++ti)
#pragma unroll
        for (int r = 0; r < 8; ++r) accO[ti][r] *= sScale[mb + r];

      for (int k0 = 0; k0 < TK; k0 += 32) {
        v16bf a = frag_a(&sP[0][0], TK, 0, k0, lane);
#pragma unroll
        for (int ti = 0; ti < 8; ++ti) {
          v16bf bf = frag_b(&sVT[cur][0][0], TK, cw + ti * 16, k0, lane); // [N=c][K=t]
          accO[ti] = wmma_bf16(a, bf, accO[ti]);
        }
      }
    }
    __syncthreads();   // consumers done; next iter may overwrite other buffer
  }

  // Epilogue: out = qin + O / l   (qin already in d_out).
  const int n = lane & 15, mb = (lane & 16) ? 8 : 0;
  float* ob = io + ((size_t)b * TSEQ + q0) * CDIM;
#pragma unroll
  for (int ti = 0; ti < 8; ++ti) {
    int col = cw + ti * 16 + n;
#pragma unroll
    for (int r = 0; r < 8; ++r) {
      int rr = mb + r;
      size_t idx = (size_t)rr * CDIM + col;
      ob[idx] = ob[idx] + accO[ti][r] / sL[rr];
    }
  }
}

// ---------------------------------------------------------------------------
// Launch. Workspace (bf16 elems): qcvt|kcvt (16M ea), 3x wT (256K ea),
// qbf|kinT|vT (16M ea)  => ~162 MB.
// ---------------------------------------------------------------------------
extern "C" void kernel_launch(void* const* d_in, const int* in_sizes, int n_in,
                              void* d_out, int out_size, void* d_ws, size_t ws_size,
                              hipStream_t stream) {
  (void)in_sizes; (void)n_in; (void)out_size; (void)ws_size;
  const float* q  = (const float*)d_in[0];
  const float* k  = (const float*)d_in[1];
  const float* Wq = (const float*)d_in[2];
  const float* bq = (const float*)d_in[3];
  const float* Wk = (const float*)d_in[4];
  const float* bk = (const float*)d_in[5];
  const float* Wv = (const float*)d_in[6];
  const float* bv = (const float*)d_in[7];
  float* out = (float*)d_out;

  const size_t nElem = (size_t)MROWS * CDIM;          // 16M
  __bf16* qcvt = (__bf16*)d_ws;
  __bf16* kcvt = qcvt + nElem;
  __bf16* wqT  = kcvt + nElem;
  __bf16* wkT  = wqT + (size_t)CDIM * CDIM;
  __bf16* wvT  = wkT + (size_t)CDIM * CDIM;
  __bf16* qbf  = wvT + (size_t)CDIM * CDIM;
  __bf16* kinT = qbf + nElem;
  __bf16* vTb  = kinT + nElem;

  cvt_kernel<<<(unsigned)(nElem / (8 * 256)), 256, 0, stream>>>(q, qcvt);
  cvt_kernel<<<(unsigned)(nElem / (8 * 256)), 256, 0, stream>>>(k, kcvt);
  wt_kernel<<<(CDIM * CDIM) / 256, 256, 0, stream>>>(Wq, wqT);
  wt_kernel<<<(CDIM * CDIM) / 256, 256, 0, stream>>>(Wk, wkT);
  wt_kernel<<<(CDIM * CDIM) / 256, 256, 0, stream>>>(Wv, wvT);

  dim3 pgrid(MROWS / 64, CDIM / 64);   // (512, 8)
  proj_kernel<<<pgrid, 128, 0, stream>>>(qcvt, wqT, bq, qbf,  out,    0);
  proj_kernel<<<pgrid, 128, 0, stream>>>(kcvt, wkT, bk, kinT, nullptr, 1);
  proj_kernel<<<pgrid, 128, 0, stream>>>(kcvt, wvT, bv, vTb,  nullptr, 2);

  attn_kernel<<<dim3(TSEQ / 16, BATCH), 128, 0, stream>>>(qbf, kinT, vTb, out);
}
